// GroupedQueryAttention_14164802142806
// MI455X (gfx1250) — compile-verified, matrix operand order checked by reading, not run on hardware
//
#include <hip/hip_runtime.h>
#include <hip/hip_bf16.h>
#include <math.h>

#define DEV __device__ __forceinline__

typedef __bf16 bf16;
typedef __attribute__((ext_vector_type(16))) __bf16 v16bf;
typedef __attribute__((ext_vector_type(8)))  float  v8f;

constexpr int Bc  = 2;
constexpr int Sc  = 2048;
constexpr int Dc  = 2048;
constexpr int HQ  = 16;
constexpr int HKV = 4;
constexpr int HD  = 128;
constexpr int KVD = HKV * HD;   // 512
constexpr int Mr  = Bc * Sc;    // 4096 rows total
constexpr float SCALE = 0.08838834764831845f; // 1/sqrt(128)

// ---------------------------------------------------------------------------
// Fragment loader: A/B 16x32 bf16 fragment per ISA 7.12.2.
// lane: m = lane&15, half = lane>>4.
// VGPR0..3 hold k = 8*half + 0..7 (contiguous 16B), VGPR4..7 hold k = 16 + 8*half + 0..7.
// ---------------------------------------------------------------------------
DEV v16bf load_frag(const bf16* src, int ld) {
    const int lane = threadIdx.x & 31;
    const int m    = lane & 15;
    const int half = lane >> 4;
    union { uint4 u[2]; v16bf v; } r;
    r.u[0] = *(const uint4*)(src + (size_t)m * ld + 8 * half);
    r.u[1] = *(const uint4*)(src + (size_t)m * ld + 16 + 8 * half);
    return r.v;
}

DEV v8f wmma_bf16(v16bf a, v16bf b, v8f c) {
    return __builtin_amdgcn_wmma_f32_16x16x32_bf16(
        false, a, false, b, (short)0, c, false, false);
}

// Async global->LDS copy: 4 x 16B per call. INST_OFFSET is added to both the
// LDS and global addresses (ISA 08_async_tensor §4.4), so one (ldsaddr, voff)
// pair covers 64 contiguous bytes. GVS mode: mem = SADDR64 + VADDR32 + IOFFSET.
DEV void async_ldg_b128x4(unsigned lds, unsigned voff, const bf16* base) {
    asm volatile(
        "global_load_async_to_lds_b128 %0, %1, %2 offset:0\n\t"
        "global_load_async_to_lds_b128 %0, %1, %2 offset:16\n\t"
        "global_load_async_to_lds_b128 %0, %1, %2 offset:32\n\t"
        "global_load_async_to_lds_b128 %0, %1, %2 offset:48"
        :: "v"(lds), "v"(voff), "s"(base) : "memory");
}

DEV void wait_async0() {
    asm volatile("s_wait_asynccnt 0x0" ::: "memory");
}

// LDS byte address = low 32 bits of the flat pointer (LDS aperture: addr[31:0]).
DEV unsigned lds_addr(const void* p) {
    return (unsigned)(uintptr_t)p;
}

// ---------------------------------------------------------------------------
// Elementwise f32 -> bf16 conversion
// ---------------------------------------------------------------------------
__global__ void cvt_f32_bf16(const float* __restrict__ src, bf16* __restrict__ dst, int n) {
    int i = blockIdx.x * blockDim.x + threadIdx.x;
    if (i < n) dst[i] = (bf16)src[i];
}

// ---------------------------------------------------------------------------
// GEMM: C[M,N] = A[M,K] * W[N,K]^T   (A, W bf16; C f32 or bf16)
// Block tile 128x128, BK=64, double-buffered LDS filled by async-to-LDS DMA.
// 8 waves: wave = (wm 0..1)x(wn 0..3), each owns a 64x32 patch (8 accumulators).
// ---------------------------------------------------------------------------
template<bool OUT_BF16>
__global__ __launch_bounds__(256) void gemm_bf16_kernel(
        const bf16* __restrict__ A, const bf16* __restrict__ W,
        void* __restrict__ Cout, int Mdim, int Ndim, int Kdim) {
    __shared__ __align__(16) bf16 As[2][128][64];
    __shared__ __align__(16) bf16 Bs[2][128][64];

    const int tid  = threadIdx.x;
    const int lane = tid & 31;
    const int wid  = tid >> 5;
    const int bm   = blockIdx.y * 128;
    const int bn   = blockIdx.x * 128;
    const int wm   = (wid >> 2) * 64;
    const int wn   = (wid & 3) * 32;

    v8f acc[4][2];
    #pragma unroll
    for (int mt = 0; mt < 4; ++mt)
        #pragma unroll
        for (int nt = 0; nt < 2; ++nt)
            #pragma unroll
            for (int j = 0; j < 8; ++j) acc[mt][nt][j] = 0.0f;

    // staging map: each thread owns 64B of A-tile and 64B of B-tile
    const int r  = tid >> 1;          // 0..127
    const int cb = (tid & 1) * 32;    // half-element column base (0 or 32)

    auto issue = [&](int buf, int k0) {
        unsigned va = (unsigned)((((size_t)(bm + r)) * Kdim + (k0 + cb)) * 2);
        async_ldg_b128x4(lds_addr(&As[buf][r][cb]), va, A);
        unsigned vb = (unsigned)((((size_t)(bn + r)) * Kdim + (k0 + cb)) * 2);
        async_ldg_b128x4(lds_addr(&Bs[buf][r][cb]), vb, W);
    };

    issue(0, 0);
    const int T = Kdim / 64;
    for (int t = 0; t < T; ++t) {
        wait_async0();                 // own tile-t loads complete (in-order)
        __syncthreads();               // tile-t visible; buf (t+1)&1 free to overwrite
        if (t + 1 < T) issue((t + 1) & 1, (t + 1) * 64);
        const int buf = t & 1;
        #pragma unroll
        for (int ks = 0; ks < 2; ++ks) {
            v16bf bfr[2];
            #pragma unroll
            for (int nt = 0; nt < 2; ++nt)
                bfr[nt] = load_frag(&Bs[buf][wn + nt * 16][ks * 32], 64);
            #pragma unroll
            for (int mt = 0; mt < 4; ++mt) {
                v16bf af = load_frag(&As[buf][wm + mt * 16][ks * 32], 64);
                #pragma unroll
                for (int nt = 0; nt < 2; ++nt)
                    acc[mt][nt] = wmma_bf16(af, bfr[nt], acc[mt][nt]);
            }
        }
    }

    const int half = lane >> 4, nl = lane & 15;
    #pragma unroll
    for (int mt = 0; mt < 4; ++mt)
        #pragma unroll
        for (int nt = 0; nt < 2; ++nt)
            #pragma unroll
            for (int j = 0; j < 8; ++j) {
                int row = bm + wm + mt * 16 + j + 8 * half;
                int col = bn + wn + nt * 16 + nl;
                float v = acc[mt][nt][j];
                if (OUT_BF16)
                    ((bf16*)Cout)[(size_t)row * Ndim + col] = (bf16)v;
                else
                    ((float*)Cout)[(size_t)row * Ndim + col] = v;
            }
}

// ---------------------------------------------------------------------------
// In-place RoPE on bf16 buffer laid out [Mr, nheads*HD].
// One thread per (row, head, j<64) pair; rotate-half formulation.
// ---------------------------------------------------------------------------
__global__ void rope_kernel(bf16* __restrict__ q, int width, int nheads) {
    int idx = blockIdx.x * blockDim.x + threadIdx.x;
    int total = Mr * nheads * 64;
    if (idx >= total) return;
    int j   = idx & 63;
    int h   = (idx >> 6) % nheads;
    int row = idx / (64 * nheads);
    int pos = row & (Sc - 1);
    float freq = __expf(-((float)(2 * j) / (float)HD) * __logf(10000.0f));
    float ang  = (float)pos * freq;
    float s, c;
    __sincosf(ang, &s, &c);
    bf16* p = q + (size_t)row * width + h * HD;
    float x1 = (float)p[j];
    float x2 = (float)p[j + 64];
    p[j]      = (bf16)(x1 * c - x2 * s);
    p[j + 64] = (bf16)(x2 * c + x1 * s);
}

// ---------------------------------------------------------------------------
// Flash attention (causal). Grid: (B*HQ, S/128). Block: 256 threads = 8 waves.
// Each wave owns 16 query rows; iterates over 32-key tiles staged in LDS.
// ---------------------------------------------------------------------------
__global__ __launch_bounds__(256) void attn_kernel(
        const bf16* __restrict__ Q, const bf16* __restrict__ Kb,
        const bf16* __restrict__ Vb, bf16* __restrict__ O) {
    __shared__ __align__(16) bf16 Kt[32][128];   // keys x head_dim
    __shared__ __align__(16) bf16 Vt[128][32];   // head_dim x keys (transposed)
    __shared__ __align__(16) bf16 Pl[8][16][32]; // wave-private P staging

    const int tid  = threadIdx.x;
    const int lane = tid & 31;
    const int wid  = tid >> 5;
    const int bh   = blockIdx.x;          // 0..31
    const int b    = bh >> 4;
    const int h    = bh & 15;
    const int hkv  = h >> 2;              // 4 Q heads per KV head
    const int qbase = blockIdx.y * 128;
    const int qrow0 = qbase + wid * 16;   // wave's first query row (within batch)
    const int half = lane >> 4, nl = lane & 15;

    // Q fragments held in registers: 16 rows x 128 dims = 4 K-chunks
    v16bf qf[4];
    const bf16* qptr = Q + (size_t)(b * Sc + qrow0) * Dc + h * HD;
    #pragma unroll
    for (int kc = 0; kc < 4; ++kc) qf[kc] = load_frag(qptr + kc * 32, Dc);

    float mi[8], li[8];
    v8f acc[8];
    #pragma unroll
    for (int j = 0; j < 8; ++j) { mi[j] = -INFINITY; li[j] = 0.0f; }
    #pragma unroll
    for (int dt = 0; dt < 8; ++dt)
        #pragma unroll
        for (int j = 0; j < 8; ++j) acc[dt][j] = 0.0f;

    const int ntiles = (qbase + 128) / 32;   // causal bound for this block
    for (int t = 0; t < ntiles; ++t) {
        const int kt = t * 32;
        __syncthreads();
        {
            int r = tid >> 3;             // 0..31 key row
            int c = (tid & 7) * 16;       // 16-half chunk of head_dim
            const uint4* gk = (const uint4*)(Kb + (size_t)(b * Sc + kt + r) * KVD + hkv * HD + c);
            *(uint4*)&Kt[r][c]     = gk[0];
            *(uint4*)&Kt[r][c + 8] = gk[1];
            const bf16* gv = Vb + (size_t)(b * Sc + kt + r) * KVD + hkv * HD + c;
            #pragma unroll
            for (int e = 0; e < 16; ++e) Vt[c + e][r] = gv[e];
        }
        __syncthreads();

        if (kt <= qrow0 + 15) {
            // S = Q * K^T : two 16-col tiles of keys
            v8f sc[2];
            #pragma unroll
            for (int nt = 0; nt < 2; ++nt)
                #pragma unroll
                for (int j = 0; j < 8; ++j) sc[nt][j] = 0.0f;
            #pragma unroll
            for (int kc = 0; kc < 4; ++kc) {
                #pragma unroll
                for (int nt = 0; nt < 2; ++nt)
                    sc[nt] = wmma_bf16(qf[kc], load_frag(&Kt[nt * 16][kc * 32], 128), sc[nt]);
            }

            // scale + causal mask + per-row tile max
            float mtile[8];
            #pragma unroll
            for (int j = 0; j < 8; ++j) {
                int qr = qrow0 + j + 8 * half;
                float v0 = sc[0][j] * SCALE;
                float v1 = sc[1][j] * SCALE;
                if (kt + nl > qr)      v0 = -1e30f;
                if (kt + 16 + nl > qr) v1 = -1e30f;
                sc[0][j] = v0; sc[1][j] = v1;
                mtile[j] = fmaxf(v0, v1);
            }
            #pragma unroll
            for (int mk = 1; mk < 16; mk <<= 1)
                #pragma unroll
                for (int j = 0; j < 8; ++j)
                    mtile[j] = fmaxf(mtile[j], __shfl_xor(mtile[j], mk, 32));

            // online softmax update
            float alpha[8], rs[8];
            #pragma unroll
            for (int j = 0; j < 8; ++j) {
                float mn = fmaxf(mi[j], mtile[j]);
                alpha[j] = __expf(mi[j] - mn);
                mi[j]    = mn;
                float p0 = __expf(sc[0][j] - mn);
                float p1 = __expf(sc[1][j] - mn);
                Pl[wid][j + 8 * half][nl]      = (bf16)p0;
                Pl[wid][j + 8 * half][16 + nl] = (bf16)p1;
                rs[j] = p0 + p1;
            }
            #pragma unroll
            for (int mk = 1; mk < 16; mk <<= 1)
                #pragma unroll
                for (int j = 0; j < 8; ++j)
                    rs[j] += __shfl_xor(rs[j], mk, 32);
            #pragma unroll
            for (int j = 0; j < 8; ++j) li[j] = li[j] * alpha[j] + rs[j];
            #pragma unroll
            for (int dt = 0; dt < 8; ++dt)
                #pragma unroll
                for (int j = 0; j < 8; ++j) acc[dt][j] *= alpha[j];

            // O += P * V  (P reloaded from LDS in A-layout; Vt gives B-layout)
            v16bf pa = load_frag(&Pl[wid][0][0], 32);
            #pragma unroll
            for (int dt = 0; dt < 8; ++dt)
                acc[dt] = wmma_bf16(pa, load_frag(&Vt[dt * 16][0], 32), acc[dt]);
        }
    }

    // normalize and write [b, s, h*HD + d] as bf16
    bf16* optr = O + (size_t)(b * Sc + qrow0) * Dc + h * HD;
    #pragma unroll
    for (int dt = 0; dt < 8; ++dt)
        #pragma unroll
        for (int j = 0; j < 8; ++j) {
            float v = acc[dt][j] / li[j];
            optr[(size_t)(j + 8 * half) * Dc + dt * 16 + nl] = (bf16)v;
        }
}

// ---------------------------------------------------------------------------
// Host launcher
// ---------------------------------------------------------------------------
extern "C" void kernel_launch(void* const* d_in, const int* in_sizes, int n_in,
                              void* d_out, int out_size, void* d_ws, size_t ws_size,
                              hipStream_t stream) {
    const float* x  = (const float*)d_in[0];
    const float* Wq = (const float*)d_in[1];
    const float* Wk = (const float*)d_in[2];
    const float* Wv = (const float*)d_in[3];
    const float* Wo = (const float*)d_in[4];

    char* ws = (char*)d_ws;
    size_t off = 0;
    auto alloc = [&](size_t bytes) -> void* {
        void* p = ws + off;
        off = (off + bytes + 255) & ~(size_t)255;
        return p;
    };

    bf16* xb  = (bf16*)alloc((size_t)Mr * Dc * 2);
    bf16* wqb = (bf16*)alloc((size_t)Dc * Dc * 2);
    bf16* wkb = (bf16*)alloc((size_t)KVD * Dc * 2);
    bf16* wvb = (bf16*)alloc((size_t)KVD * Dc * 2);
    bf16* wob = (bf16*)alloc((size_t)Dc * Dc * 2);
    bf16* qb  = (bf16*)alloc((size_t)Mr * Dc * 2);
    bf16* kb  = (bf16*)alloc((size_t)Mr * KVD * 2);
    bf16* vb  = (bf16*)alloc((size_t)Mr * KVD * 2);
    bf16* ab  = (bf16*)alloc((size_t)Mr * Dc * 2);

    auto cvt = [&](const float* s, bf16* dp, int n) {
        cvt_f32_bf16<<<(n + 255) / 256, 256, 0, stream>>>(s, dp, n);
    };
    cvt(x,  xb,  Mr * Dc);
    cvt(Wq, wqb, Dc * Dc);
    cvt(Wk, wkb, KVD * Dc);
    cvt(Wv, wvb, KVD * Dc);
    cvt(Wo, wob, Dc * Dc);

    dim3 blk(256);
    // Q, K, V projections (bf16 outputs)
    gemm_bf16_kernel<true><<<dim3(Dc / 128, Mr / 128), blk, 0, stream>>>(xb, wqb, qb, Mr, Dc, Dc);
    gemm_bf16_kernel<true><<<dim3(KVD / 128, Mr / 128), blk, 0, stream>>>(xb, wkb, kb, Mr, KVD, Dc);
    gemm_bf16_kernel<true><<<dim3(KVD / 128, Mr / 128), blk, 0, stream>>>(xb, wvb, vb, Mr, KVD, Dc);

    // RoPE on Q and K (in place)
    rope_kernel<<<(Mr * HQ * 64 + 255) / 256, 256, 0, stream>>>(qb, Dc, HQ);
    rope_kernel<<<(Mr * HKV * 64 + 255) / 256, 256, 0, stream>>>(kb, KVD, HKV);

    // Flash attention
    attn_kernel<<<dim3(Bc * HQ, Sc / 128), blk, 0, stream>>>(qb, kb, vb, ab);

    // Output projection -> f32 d_out
    gemm_bf16_kernel<false><<<dim3(Dc / 128, Mr / 128), blk, 0, stream>>>(ab, wob, (float*)d_out, Mr, Dc, Dc);
}
